// ParallelAttntn_83691732730424
// MI455X (gfx1250) — matrix operand order repaired and checked by reading, not compiled, not run on hardware
//
#include <hip/hip_runtime.h>
#include <hip/hip_bf16.h>
#include <math.h>

// ---------------------------------------------------------------------------
// XLNet two-stream relative attention for MI455X (gfx1250, wave32, WMMA).
// GEMMs: v_wmma_f32_16x16x32_bf16, tiles DMA'd into LDS by the Tensor Data
// Mover (tensor_load_to_lds, TENSORcnt) and repacked f32->bf16 on the VALU.
// Softmax / LayerNorm / one-hot scatter are VALU glue.
// ---------------------------------------------------------------------------

#define HID   1024
#define NH    8
#define DH    1024
#define NDH   8192      // NH*DH
#define BB    2
#define QL    512
#define ML    512
#define KL    1024      // ML+QL
#define RL    1025      // KL+1
#define NP    128

#if __has_builtin(__builtin_amdgcn_tensor_load_to_lds)
#define USE_TDM 1
#else
#define USE_TDM 0
#endif

typedef __attribute__((ext_vector_type(16))) __bf16 bf16x16;
typedef __attribute__((ext_vector_type(8)))  float  floatx8;
typedef __attribute__((ext_vector_type(4)))  unsigned int uint32x4;
typedef __attribute__((ext_vector_type(8)))  int  int32x8;
typedef __attribute__((ext_vector_type(4)))  int  int32x4;

struct Chunk32 { uint4 a; uint4 b; };   // 32 bytes = one per-lane WMMA fragment

static __device__ __forceinline__ unsigned short f2bf(float f) {
    unsigned u = __builtin_bit_cast(unsigned, f);
    u += 0x7FFFu + ((u >> 16) & 1u);          // round-to-nearest-even
    return (unsigned short)(u >> 16);
}

static __device__ __forceinline__ bf16x16 load_frag_a(const unsigned short* rowp, int hi) {
    Chunk32 c;
    c.a = *(const uint4*)(rowp + hi * 8);        // K = hi*8 + 0..7
    c.b = *(const uint4*)(rowp + hi * 8 + 16);   // K = hi*8 + 16..23
    return __builtin_bit_cast(bf16x16, c);
}
static __device__ __forceinline__ bf16x16 load_frag_b(const unsigned short* rowp, int hi) {
    Chunk32 c;
    c.a = *(const uint4*)(rowp + hi * 8);        // K = hi*8 + 0..7
    c.b = *(const uint4*)(rowp + hi * 8 + 16);   // K = hi*8 + 16..23
    return __builtin_bit_cast(bf16x16, c);
}

#if USE_TDM
// Issue one TDM 2D-tile load (f32 elements) global -> LDS.
//  lds_off   : destination byte offset inside the workgroup LDS
//  gptr      : tile start in global memory
//  tile_cols : contiguous elements per tile row (dim0)
//  tile_rows : tile rows (dim1)
//  ld        : row stride in elements (tensor_dim0_stride)
//  rem_cols / rem_rows : remaining valid tensor extent from the tile start
//                        (OOB rows/cols read as zero per D# semantics)
static __device__ __forceinline__ void tdm_load_tile_f32(
    unsigned lds_off, const float* gptr,
    int tile_cols, int tile_rows, int ld,
    int rem_cols, int rem_rows)
{
    unsigned long long ga = (unsigned long long)(uintptr_t)gptr;
    uint32x4 g0;
    g0[0] = 1u;                                               // count=1, user mode
    g0[1] = lds_off;                                          // lds_addr (bytes)
    g0[2] = (unsigned)(ga & 0xFFFFFFFFull);                   // global_addr[31:0]
    g0[3] = (unsigned)((ga >> 32) & 0x01FFFFFFull)            // global_addr[56:32]
          | (2u << 30);                                       // type = 2 ("image")
    int32x8 g1;
    g1[0] = (2 << 16);                                        // data_size: 4 bytes
    g1[1] = (rem_cols & 0xFFFF) << 16;                        // tensor_dim0 lo16
    g1[2] = ((rem_cols >> 16) & 0xFFFF)                       // tensor_dim0 hi16
          | ((rem_rows & 0xFFFF) << 16);                      // tensor_dim1 lo16
    g1[3] = ((rem_rows >> 16) & 0xFFFF)                       // tensor_dim1 hi16
          | (tile_cols << 16);                                // tile_dim0
    g1[4] = (tile_rows & 0xFFFF);                             // tile_dim1, tile_dim2=0
    g1[5] = ld;                                               // tensor_dim0_stride lo32
    g1[6] = 0;                                                // stride hi / dim1_stride lo
    g1[7] = 0;
    int32x4 z4 = {0, 0, 0, 0};
    int32x8 z8 = {0, 0, 0, 0, 0, 0, 0, 0};
    // amdgpu-toolchain (clang-23) 6-arg form: (g0, g1, g2, g3, g_ext, cpol)
    __builtin_amdgcn_tensor_load_to_lds(g0, g1, z4, z4, z8, 0);
}
#endif

// ---------------------------------------------------------------------------
// Generic batched GEMM: C[b] = A[b] * op(B[b]) (+ bias)
//   A: M x K, row stride lda, batch stride sA     (f32)
//   B: if transB: stored [N x K] ldb ; else [K x N] ldb
//   C: M x N, row stride ldc, batch stride sC     (f32)
// K must be a multiple of 32. M, N arbitrary (guarded).
// Block: 256 threads (8 waves), tile 128x128x32, wave tile 64x32 (4x2 WMMAs).
// ---------------------------------------------------------------------------
__global__ __launch_bounds__(256) void gemm_bf16_wmma(
    const float* __restrict__ A, int lda, long long sA,
    const float* __restrict__ Bm, int ldb, long long sB, int transB,
    float* __restrict__ C, int ldc, long long sC,
    int M, int N, int K, const float* __restrict__ bias)
{
    __shared__ __align__(16) unsigned short As[128][40];   // bf16 WMMA staging
    __shared__ __align__(16) unsigned short Bs[128][40];
#if USE_TDM
    __shared__ __align__(16) float Araw[128 * 32];         // TDM landing pads (f32)
    __shared__ __align__(16) float Braw[128 * 32];
#endif

    const int tid  = threadIdx.x;
    const int lane = tid & 31;
    const int wave = tid >> 5;
    const int lo   = lane & 15;
    const int hi   = lane >> 4;
    const int wr   = (wave >> 2) * 64;   // wave row offset: 0 / 64
    const int wc   = (wave & 3) * 32;    // wave col offset: 0/32/64/96

    const long long batch = blockIdx.z;
    A  += batch * sA;
    Bm += batch * sB;
    C  += batch * sC;
    const int bm = blockIdx.y * 128;
    const int bn = blockIdx.x * 128;

    floatx8 zero;
    #pragma unroll
    for (int v = 0; v < 8; ++v) zero[v] = 0.0f;
    floatx8 acc[4][2];
    #pragma unroll
    for (int i = 0; i < 4; ++i)
        #pragma unroll
        for (int j = 0; j < 2; ++j) acc[i][j] = zero;

    for (int k0 = 0; k0 < K; k0 += 32) {
#if USE_TDM
        // ---- DMA the two f32 tiles into LDS via the Tensor Data Mover ----
        if (wave == 0) {
            tdm_load_tile_f32((unsigned)(uintptr_t)(void*)&Araw[0],
                              A + (long long)bm * lda + k0,
                              32, 128, lda, K - k0, M - bm);
            if (transB) {
                // tile rows = output cols, contiguous along K
                tdm_load_tile_f32((unsigned)(uintptr_t)(void*)&Braw[0],
                                  Bm + (long long)bn * ldb + k0,
                                  32, 128, ldb, K - k0, N - bn);
            } else {
                // tile rows = K, contiguous along output cols
                tdm_load_tile_f32((unsigned)(uintptr_t)(void*)&Braw[0],
                                  Bm + (long long)k0 * ldb + bn,
                                  128, 32, ldb, N - bn, K - k0);
            }
            __builtin_amdgcn_s_wait_tensorcnt(0);
        }
        __syncthreads();   // publish TDM-written LDS to all waves

        // ---- repack f32 -> bf16 into the padded WMMA tiles ----
        #pragma unroll
        for (int it = 0; it < 4; ++it) {
            int idx = tid + it * 256;
            int r   = idx >> 3;
            int cq  = (idx & 7) << 2;
            float4 f = *(const float4*)&Araw[r * 32 + cq];
            As[r][cq + 0] = f2bf(f.x); As[r][cq + 1] = f2bf(f.y);
            As[r][cq + 2] = f2bf(f.z); As[r][cq + 3] = f2bf(f.w);
        }
        if (transB) {
            #pragma unroll
            for (int it = 0; it < 4; ++it) {
                int idx = tid + it * 256;
                int r   = idx >> 3;
                int cq  = (idx & 7) << 2;
                float4 f = *(const float4*)&Braw[r * 32 + cq];
                Bs[r][cq + 0] = f2bf(f.x); Bs[r][cq + 1] = f2bf(f.y);
                Bs[r][cq + 2] = f2bf(f.z); Bs[r][cq + 3] = f2bf(f.w);
            }
        } else {
            #pragma unroll
            for (int it = 0; it < 4; ++it) {
                int idx = tid + it * 256;
                int nn  = idx & 127;
                int kg  = (idx >> 7) << 2;
                Bs[nn][kg + 0] = f2bf(Braw[(kg + 0) * 128 + nn]);
                Bs[nn][kg + 1] = f2bf(Braw[(kg + 1) * 128 + nn]);
                Bs[nn][kg + 2] = f2bf(Braw[(kg + 2) * 128 + nn]);
                Bs[nn][kg + 3] = f2bf(Braw[(kg + 3) * 128 + nn]);
            }
        }
#else
        // ---- fallback: per-thread global loads + convert ----
        #pragma unroll
        for (int it = 0; it < 4; ++it) {
            int idx  = tid + it * 256;
            int r    = idx >> 3;
            int cq   = (idx & 7) << 2;
            int grow = bm + r;
            float4 f = make_float4(0.f, 0.f, 0.f, 0.f);
            if (grow < M) {
                const float* ap = A + (long long)grow * lda + k0 + cq;
                f = *(const float4*)ap;
                if (k0 + 32 < K) __builtin_prefetch(ap + 32, 0, 1);
            }
            As[r][cq + 0] = f2bf(f.x); As[r][cq + 1] = f2bf(f.y);
            As[r][cq + 2] = f2bf(f.z); As[r][cq + 3] = f2bf(f.w);
        }
        if (transB) {
            #pragma unroll
            for (int it = 0; it < 4; ++it) {
                int idx = tid + it * 256;
                int r   = idx >> 3;
                int cq  = (idx & 7) << 2;
                int gn  = bn + r;
                float4 f = make_float4(0.f, 0.f, 0.f, 0.f);
                if (gn < N) f = *(const float4*)(Bm + (long long)gn * ldb + k0 + cq);
                Bs[r][cq + 0] = f2bf(f.x); Bs[r][cq + 1] = f2bf(f.y);
                Bs[r][cq + 2] = f2bf(f.z); Bs[r][cq + 3] = f2bf(f.w);
            }
        } else {
            #pragma unroll
            for (int it = 0; it < 4; ++it) {
                int idx = tid + it * 256;
                int nn  = idx & 127;
                int kg  = (idx >> 7) << 2;
                int gn  = bn + nn;
                float4 f = make_float4(0.f, 0.f, 0.f, 0.f);
                if (gn < N) {
                    const float*    bp   = Bm + (long long)(k0 + kg) * ldb + gn;
                    const long long lldb = ldb;
                    f.x = bp[0]; f.y = bp[lldb]; f.z = bp[2 * lldb]; f.w = bp[3 * lldb];
                }
                Bs[nn][kg + 0] = f2bf(f.x); Bs[nn][kg + 1] = f2bf(f.y);
                Bs[nn][kg + 2] = f2bf(f.z); Bs[nn][kg + 3] = f2bf(f.w);
            }
        }
#endif
        __syncthreads();

        bf16x16 af[4], bfr[2];
        #pragma unroll
        for (int tr = 0; tr < 4; ++tr) af[tr]  = load_frag_a(&As[wr + tr * 16 + lo][0], hi);
        #pragma unroll
        for (int tc = 0; tc < 2; ++tc) bfr[tc] = load_frag_b(&Bs[wc + tc * 16 + lo][0], hi);

        #pragma unroll
        for (int tr = 0; tr < 4; ++tr)
            #pragma unroll
            for (int tc = 0; tc < 2; ++tc)
                acc[tr][tc] = __builtin_amdgcn_wmma_f32_16x16x32_bf16(
                    false, af[tr], false, bfr[tc], (short)0, acc[tr][tc], false, false);
        __syncthreads();
    }

    // ---- writeback (C/D layout: VGPR v -> row hi*8+v, col = lo) ----
    #pragma unroll
    for (int tc = 0; tc < 2; ++tc) {
        int col = bn + wc + tc * 16 + lo;
        if (col >= N) continue;
        float bv = bias ? bias[col] : 0.0f;
        #pragma unroll
        for (int tr = 0; tr < 4; ++tr) {
            #pragma unroll
            for (int v = 0; v < 8; ++v) {
                int row = bm + wr + tr * 16 + hi * 8 + v;
                if (row < M) C[(long long)row * ldc + col] = acc[tr][tc][v] + bv;
            }
        }
    }
}

// ---------------------------------------------------------------------------
// q + per-head bias broadcast (three biases in one pass). Layout [rows][NDH].
// ---------------------------------------------------------------------------
__global__ void biasadd3_kernel(const float* __restrict__ q,
                                const float* __restrict__ b1,
                                const float* __restrict__ b2,
                                const float* __restrict__ b3,
                                float* __restrict__ o1, float* __restrict__ o2,
                                float* __restrict__ o3, int total)
{
    for (int idx = blockIdx.x * blockDim.x + threadIdx.x; idx < total;
         idx += gridDim.x * blockDim.x) {
        int nd = idx & (NDH - 1);
        float v = q[idx];
        o1[idx] = v + b1[nd];
        o2[idx] = v + b2[nd];
        o3[idx] = v + b3[nd];
    }
}

// ---------------------------------------------------------------------------
// ef[row,n,s] = dot(qs[row,n,:], seg_embed[s,n,:]) ; row = i*B+b in [0,1024)
// blockIdx.x = (row*8+n)*2+s
// ---------------------------------------------------------------------------
__global__ __launch_bounds__(256) void ef_kernel(const float* __restrict__ qs,
                                                 const float* __restrict__ seg_embed,
                                                 float* __restrict__ ef)
{
    __shared__ float red[256];
    int bid = blockIdx.x;
    int s   = bid & 1;
    int n   = (bid >> 1) & 7;
    int row = bid >> 4;
    const float* qp = qs        + ((size_t)row * NH + n) * DH;
    const float* sp = seg_embed + ((size_t)s   * NH + n) * DH;
    float a = 0.0f;
    for (int d = threadIdx.x; d < DH; d += 256) a += qp[d] * sp[d];
    red[threadIdx.x] = a;
    __syncthreads();
    for (int st = 128; st > 0; st >>= 1) {
        if (threadIdx.x < st) red[threadIdx.x] += red[threadIdx.x + st];
        __syncthreads();
    }
    if (threadIdx.x == 0) ef[bid] = red[0];
}

// ---------------------------------------------------------------------------
// score = (ac + rel_shift(bd) + seg_select(ef)) * scale - 1e30*mask ; softmax.
// grid (i=512, batch=b*8+n). rel_shift == flat read of the [512,1025] plane
// at offset i*1024 + j + 512.
// ---------------------------------------------------------------------------
__global__ __launch_bounds__(256) void score_softmax_kernel(
    const float* __restrict__ ac, const float* __restrict__ bd,
    const float* __restrict__ ef, const float* __restrict__ seg_mat,
    const float* __restrict__ mask, float* __restrict__ prob, float scale)
{
    __shared__ float red[256];
    const int i   = blockIdx.x;
    const int bi  = blockIdx.y;          // b*8+n
    const int b   = bi >> 3;
    const int n   = bi & 7;
    const int tid = threadIdx.x;
    const float* acp = ac + (size_t)bi * (QL * KL) + (size_t)i * KL;
    const float* bdp = bd + (size_t)bi * (QL * RL);
    float e0 = ef[(((size_t)i * BB + b) * NH + n) * 2 + 0];
    float e1 = ef[(((size_t)i * BB + b) * NH + n) * 2 + 1];

    float vals[4];
    float vmax = -3.4e38f;
    #pragma unroll
    for (int u = 0; u < 4; ++u) {
        int j = tid + u * 256;
        size_t ij = (size_t)i * KL + j;
        float sc = acp[j] + bdp[ij + QL];           // rel_shift
        float s0 = seg_mat[(ij * BB + b) * 2 + 0];
        float s1 = seg_mat[(ij * BB + b) * 2 + 1];
        sc += s0 * e0 + s1 * e1;
        sc = sc * scale - 1e30f * mask[ij * BB + b];
        vals[u] = sc;
        vmax = fmaxf(vmax, sc);
    }
    red[tid] = vmax;
    __syncthreads();
    for (int st = 128; st > 0; st >>= 1) {
        if (tid < st) red[tid] = fmaxf(red[tid], red[tid + st]);
        __syncthreads();
    }
    vmax = red[0];
    __syncthreads();
    float lsum = 0.0f;
    #pragma unroll
    for (int u = 0; u < 4; ++u) { vals[u] = __expf(vals[u] - vmax); lsum += vals[u]; }
    red[tid] = lsum;
    __syncthreads();
    for (int st = 128; st > 0; st >>= 1) {
        if (tid < st) red[tid] += red[tid + st];
        __syncthreads();
    }
    float inv = 1.0f / red[0];
    float* pp = prob + (size_t)bi * (QL * KL) + (size_t)i * KL;
    #pragma unroll
    for (int u = 0; u < 4; ++u) pp[tid + u * 256] = vals[u] * inv;
}

// ---------------------------------------------------------------------------
// qg[l,b,:] = sum_m tm[m,l,b] * qgm[m,b,:]   (blockIdx.x = l*2+b)
// ---------------------------------------------------------------------------
__global__ __launch_bounds__(256) void scatter_q_kernel(const float* __restrict__ qgm,
                                                        const float* __restrict__ tm,
                                                        float* __restrict__ qg)
{
    __shared__ int   nf;
    __shared__ int   ml[32];
    __shared__ float mw[32];
    int l = blockIdx.x >> 1, b = blockIdx.x & 1, tid = threadIdx.x;
    if (tid == 0) nf = 0;
    __syncthreads();
    for (int m = tid; m < NP; m += 256) {
        float w = tm[((size_t)m * QL + l) * BB + b];
        if (w != 0.0f) {
            int k = atomicAdd(&nf, 1);
            if (k < 32) { ml[k] = m; mw[k] = w; }
        }
    }
    __syncthreads();
    int cnt = nf > 32 ? 32 : nf;
    float* out = qg + ((size_t)l * BB + b) * NDH;
    for (int c = tid; c < NDH; c += 256) {
        float a = 0.0f;
        for (int k = 0; k < cnt; ++k)
            a += mw[k] * qgm[((size_t)ml[k] * BB + b) * NDH + c];
        out[c] = a;
    }
}

// ---------------------------------------------------------------------------
// avgm[m,b,:] = sum_l tm[m,l,b] * av[l,b,:]   (blockIdx.x = m*2+b)
// ---------------------------------------------------------------------------
__global__ __launch_bounds__(256) void gather_av_kernel(const float* __restrict__ av,
                                                        const float* __restrict__ tm,
                                                        float* __restrict__ avgm)
{
    __shared__ int   nf;
    __shared__ int   ll[8];
    __shared__ float lw[8];
    int m = blockIdx.x >> 1, b = blockIdx.x & 1, tid = threadIdx.x;
    if (tid == 0) nf = 0;
    __syncthreads();
    for (int l = tid; l < QL; l += 256) {
        float w = tm[((size_t)m * QL + l) * BB + b];
        if (w != 0.0f) {
            int k = atomicAdd(&nf, 1);
            if (k < 8) { ll[k] = l; lw[k] = w; }
        }
    }
    __syncthreads();
    int cnt = nf > 8 ? 8 : nf;
    float* out = avgm + ((size_t)m * BB + b) * NDH;
    for (int c = tid; c < NDH; c += 256) {
        float a = 0.0f;
        for (int k = 0; k < cnt; ++k)
            a += lw[k] * av[((size_t)ll[k] * BB + b) * NDH + c];
        out[c] = a;
    }
}

// ---------------------------------------------------------------------------
// out[row,:] = LayerNorm(o[row,:] + res[row,:]) * gamma + beta
// ---------------------------------------------------------------------------
__global__ __launch_bounds__(256) void ln_kernel(const float* __restrict__ o,
                                                 const float* __restrict__ res,
                                                 const float* __restrict__ gamma,
                                                 const float* __restrict__ beta,
                                                 float* __restrict__ out)
{
    __shared__ float red[256];
    int row = blockIdx.x, tid = threadIdx.x;
    const float* op = o   + (size_t)row * HID;
    const float* rp = res + (size_t)row * HID;
    float x[4];
    float s = 0.0f;
    #pragma unroll
    for (int u = 0; u < 4; ++u) { x[u] = op[tid + u * 256] + rp[tid + u * 256]; s += x[u]; }
    red[tid] = s;
    __syncthreads();
    for (int st = 128; st > 0; st >>= 1) {
        if (tid < st) red[tid] += red[tid + st];
        __syncthreads();
    }
    float mu = red[0] * (1.0f / HID);
    __syncthreads();
    float v = 0.0f;
    #pragma unroll
    for (int u = 0; u < 4; ++u) { float d = x[u] - mu; v += d * d; }
    red[tid] = v;
    __syncthreads();
    for (int st = 128; st > 0; st >>= 1) {
        if (tid < st) red[tid] += red[tid + st];
        __syncthreads();
    }
    float inv = rsqrtf(red[0] * (1.0f / HID) + 1e-5f);
    #pragma unroll
    for (int u = 0; u < 4; ++u) {
        int c = tid + u * 256;
        out[(size_t)row * HID + c] = gamma[c] * (x[u] - mu) * inv + beta[c];
    }
}

// ---------------------------------------------------------------------------
// Host side
// ---------------------------------------------------------------------------
static inline void launch_gemm(hipStream_t st,
                               const float* A, int lda, long long sA,
                               const float* Bm, int ldb, long long sB, int tB,
                               float* C, int ldc, long long sC,
                               int M, int N, int K, int nb, const float* bias)
{
    dim3 grid((N + 127) / 128, (M + 127) / 128, nb);
    gemm_bf16_wmma<<<grid, 256, 0, st>>>(A, lda, sA, Bm, ldb, sB, tB,
                                         C, ldc, sC, M, N, K, bias);
}

extern "C" void kernel_launch(void* const* d_in, const int* in_sizes, int n_in,
                              void* d_out, int out_size, void* d_ws, size_t ws_size,
                              hipStream_t stream)
{
    const float* h         = (const float*)d_in[0];
    const float* g         = (const float*)d_in[1];
    const float* r         = (const float*)d_in[2];
    const float* mems      = (const float*)d_in[3];
    const float* rwb       = (const float*)d_in[4];
    const float* rrb       = (const float*)d_in[5];
    const float* rsb       = (const float*)d_in[6];
    const float* seg_embed = (const float*)d_in[7];
    const float* q_proj    = (const float*)d_in[8];
    const float* k_proj    = (const float*)d_in[9];
    const float* v_proj    = (const float*)d_in[10];
    const float* r_proj    = (const float*)d_in[11];
    const float* proj_o    = (const float*)d_in[12];
    const float* Wk        = (const float*)d_in[13];
    const float* bk        = (const float*)d_in[14];
    const float* Wv        = (const float*)d_in[15];
    const float* bv        = (const float*)d_in[16];
    const float* Wq        = (const float*)d_in[17];
    const float* bq        = (const float*)d_in[18];
    const float* Wr        = (const float*)d_in[19];
    const float* br        = (const float*)d_in[20];
    const float* Wg        = (const float*)d_in[21];
    const float* bg        = (const float*)d_in[22];
    const float* Wd        = (const float*)d_in[23];
    const float* bd_b      = (const float*)d_in[24];
    const float* gamma     = (const float*)d_in[25];
    const float* beta      = (const float*)d_in[26];
    const float* seg_mat   = (const float*)d_in[27];
    const float* mask_h    = (const float*)d_in[28];
    const float* mask_g    = (const float*)d_in[29];
    const float* tm        = (const float*)d_in[30];

    float* W = (float*)d_ws;
    size_t off = 0;
    auto alloc = [&](size_t n) { size_t p = off; off += n; return p; };
    const size_t o_cat  = alloc((size_t)KL * BB * HID);         // 2048 x 1024
    const size_t o_t1   = alloc((size_t)RL * BB * NDH);         // 2050 x 8192 (stage-1 tmp)
    const size_t o_kh   = alloc((size_t)KL * BB * NH * DH);     // [t,b,n,d]
    const size_t o_vh   = alloc((size_t)KL * BB * NH * DH);
    const size_t o_kr   = alloc((size_t)RL * BB * NH * DH);
    const size_t o_qh   = alloc((size_t)QL * BB * NH * DH);
    const size_t o_qgm  = alloc((size_t)NP * BB * NH * DH);
    const size_t o_qg   = alloc((size_t)QL * BB * NH * DH);
    const size_t o_qw   = alloc((size_t)QL * BB * NH * DH);
    const size_t o_qr   = alloc((size_t)QL * BB * NH * DH);
    const size_t o_qs   = alloc((size_t)QL * BB * NH * DH);
    const size_t o_ef   = alloc((size_t)QL * BB * NH * 2);
    const size_t o_ac   = alloc((size_t)16 * QL * KL);
    const size_t o_bd   = alloc((size_t)16 * QL * RL);
    const size_t o_pr   = alloc((size_t)16 * QL * KL);
    const size_t o_av   = alloc((size_t)QL * BB * NH * DH);
    const size_t o_avgm = alloc((size_t)NP * BB * NH * DH);
    const size_t o_vb   = alloc((size_t)QL * BB * NH * DH);
    const size_t o_ob   = alloc((size_t)QL * BB * HID);
    (void)ws_size; (void)in_sizes; (void)n_in; (void)out_size;

    const float scale = 1.0f / 32.0f;   // 1/sqrt(1024)
    const int rowsBN = 16384;           // token-row stride in [t,b,n,d] layouts

    // cat = concat(mems, h) along axis 0  (both contiguous [T*B, H])
    (void)hipMemcpyAsync(W + o_cat, mems, (size_t)ML * BB * HID * 4,
                         hipMemcpyDeviceToDevice, stream);
    (void)hipMemcpyAsync(W + o_cat + (size_t)ML * BB * HID, h, (size_t)QL * BB * HID * 4,
                         hipMemcpyDeviceToDevice, stream);

    // ---- projections: X @ proj -> t1 ; t1 @ W^T + b -> heads ----
    launch_gemm(stream, W + o_cat, HID, 0, k_proj, NDH, 0, 0, W + o_t1, NDH, 0,
                KL * BB, NDH, HID, 1, nullptr);
    launch_gemm(stream, W + o_t1, DH, 0, Wk, DH, 0, 1, W + o_kh, HID, 0,
                KL * BB * NH, HID, DH, 1, bk);

    launch_gemm(stream, W + o_cat, HID, 0, v_proj, NDH, 0, 0, W + o_t1, NDH, 0,
                KL * BB, NDH, HID, 1, nullptr);
    launch_gemm(stream, W + o_t1, DH, 0, Wv, DH, 0, 1, W + o_vh, HID, 0,
                KL * BB * NH, HID, DH, 1, bv);

    launch_gemm(stream, r, HID, 0, r_proj, NDH, 0, 0, W + o_t1, NDH, 0,
                RL * BB, NDH, HID, 1, nullptr);
    launch_gemm(stream, W + o_t1, DH, 0, Wr, DH, 0, 1, W + o_kr, HID, 0,
                RL * BB * NH, HID, DH, 1, br);

    launch_gemm(stream, h, HID, 0, q_proj, NDH, 0, 0, W + o_t1, NDH, 0,
                QL * BB, NDH, HID, 1, nullptr);
    launch_gemm(stream, W + o_t1, DH, 0, Wq, DH, 0, 1, W + o_qh, HID, 0,
                QL * BB * NH, HID, DH, 1, bq);

    launch_gemm(stream, g, HID, 0, q_proj, NDH, 0, 0, W + o_t1, NDH, 0,
                NP * BB, NDH, HID, 1, nullptr);
    launch_gemm(stream, W + o_t1, DH, 0, Wg, DH, 0, 1, W + o_qgm, HID, 0,
                NP * BB * NH, HID, DH, 1, bg);

    // qg[l,b] = sum_m tm[m,l,b] * qgm[m,b]
    scatter_q_kernel<<<QL * BB, 256, 0, stream>>>(W + o_qgm, tm, W + o_qg);

    const int qtotal = QL * BB * NDH;   // 8,388,608

    for (int s = 0; s < 2; ++s) {       // s=0: h stream, s=1: g stream
        const float* qsrc = (s == 0) ? (W + o_qh) : (W + o_qg);
        const float* msk  = (s == 0) ? mask_h : mask_g;

        biasadd3_kernel<<<4096, 256, 0, stream>>>(qsrc, rwb, rrb, rsb,
                                                  W + o_qw, W + o_qr, W + o_qs, qtotal);

        // ac[i,j] = (q+rwb) . kh   (16 batches over b*8+n)
        launch_gemm(stream, W + o_qw, rowsBN, DH, W + o_kh, rowsBN, DH, 1,
                    W + o_ac, KL, (long long)QL * KL, QL, KL, DH, 16, nullptr);
        // bd[i,j'] = (q+rrb) . kr  (N = RL = 1025)
        launch_gemm(stream, W + o_qr, rowsBN, DH, W + o_kr, rowsBN, DH, 1,
                    W + o_bd, RL, (long long)QL * RL, QL, RL, DH, 16, nullptr);
        // ef dots
        ef_kernel<<<QL * BB * NH * 2, 256, 0, stream>>>(W + o_qs, seg_embed, W + o_ef);
        // softmax
        score_softmax_kernel<<<dim3(QL, 16), 256, 0, stream>>>(
            W + o_ac, W + o_bd, W + o_ef, seg_mat, msk, W + o_pr, scale);
        // av = prob @ vh
        launch_gemm(stream, W + o_pr, KL, (long long)QL * KL, W + o_vh, rowsBN, DH, 0,
                    W + o_av, rowsBN, DH, QL, DH, KL, 16, nullptr);

        const float* avp = W + o_av;
        int rows = QL * BB;             // output token rows
        const float* resid = h;
        float* outp = (float*)d_out;
        if (s == 1) {
            gather_av_kernel<<<NP * BB, 256, 0, stream>>>(W + o_av, tm, W + o_avgm);
            avp   = W + o_avgm;
            rows  = NP * BB;
            resid = g;
            outp  = (float*)d_out + (size_t)QL * BB * HID;
        }
        // v = av @ Wd^T + bd
        launch_gemm(stream, avp, DH, 0, Wd, DH, 0, 1, W + o_vb, HID, 0,
                    rows * NH, HID, DH, 1, bd_b);
        // o = v @ proj_o^T
        launch_gemm(stream, W + o_vb, NDH, 0, proj_o, NDH, 0, 1, W + o_ob, HID, 0,
                    rows, HID, NDH, 1, nullptr);
        // LayerNorm(residual + o)
        ln_kernel<<<rows, 256, 0, stream>>>(W + o_ob, resid, gamma, beta, outp);
    }
}
